// Attention_54150947668207
// MI455X (gfx1250) — matrix operand order repaired and checked by reading, not compiled, not run on hardware
//
#include <hip/hip_runtime.h>
#include <hip/hip_bf16.h>

#define DIM   384
#define HEADS 8
#define CH    48          // channels per head
#define CP    64          // padded channels per head (K-dim padding for WMMA)
#define B2    2
#define HH    128
#define WW    128
#define HW    16384
#define BH    16          // B2*HEADS

typedef __attribute__((ext_vector_type(16))) _Float16 v16h;
typedef __attribute__((ext_vector_type(8)))  _Float16 v8h;
typedef __attribute__((ext_vector_type(8)))  float    v8f;

union F16x16 { v16h v; v8h h[2]; };

// ---------------- workspace layout (bytes) ----------------
#define OFF_QH   0ull                                   // [B2*DIM][HW] f16
#define OFF_KH   (OFF_QH   + 2ull*DIM*HW*2)             // [B2*DIM][HW] f16
#define OFF_VT   (OFF_KH   + 2ull*DIM*HW*2)             // [BH][HW][CP] f16 (transposed, padded)
#define OFF_OUTT (OFF_VT   + (unsigned long long)BH*HW*CP*2)   // [B2][HW][DIM] f16 (transposed)
#define OFF_ATTN (OFF_OUTT + 2ull*HW*DIM*2)             // [BH][48][48] f32
#define OFF_M    (OFF_ATTN + (unsigned long long)BH*CH*CH*4)   // [BH][48][64] f16
#define OFF_NQ   (OFF_M    + (unsigned long long)BH*CH*CP*2)   // [768] f32 sumsq(q)
#define OFF_NK   (OFF_NQ   + 768ull*4)
#define OFF_WP   (OFF_NK   + 768ull*4)                  // [384][384] f16
#define WS_NEED  (OFF_WP   + (unsigned long long)DIM*DIM*2)

// ---------------- wproj f32 -> f16 ----------------
__global__ void k_cvt_wproj(const float* __restrict__ w, _Float16* __restrict__ w16) {
    int i = blockIdx.x * 256 + threadIdx.x;
    if (i < DIM * DIM) w16[i] = (_Float16)w[i];
}

// ---------------- depthwise 3x3 for q/k: f16 out + fused sum-of-squares ----------------
// grid (HW/256, DIM, B2), block 256
__global__ void k_dwconv_qk(const float* __restrict__ src, const float* __restrict__ w9,
                            _Float16* __restrict__ dst, float* __restrict__ normsq) {
    int b = blockIdx.z, c = blockIdx.y;
    int n = blockIdx.x * 256 + threadIdx.x;
    int y = n >> 7, x = n & 127;
    const float* sp = src + ((size_t)(b * DIM + c)) * HW;
    const float* wp = w9 + c * 9;
    float acc = 0.f;
    #pragma unroll
    for (int ky = 0; ky < 3; ++ky) {
        int yy = y + ky - 1;
        if (yy < 0 || yy >= HH) continue;
        #pragma unroll
        for (int kx = 0; kx < 3; ++kx) {
            int xx = x + kx - 1;
            if (xx < 0 || xx >= WW) continue;
            acc += sp[yy * WW + xx] * wp[ky * 3 + kx];
        }
    }
    dst[((size_t)(b * DIM + c)) * HW + n] = (_Float16)acc;
    __shared__ float red[256];
    red[threadIdx.x] = acc * acc;
    __syncthreads();
    for (int s = 128; s > 0; s >>= 1) {
        if (threadIdx.x < s) red[threadIdx.x] += red[threadIdx.x + s];
        __syncthreads();
    }
    if (threadIdx.x == 0) atomicAdd(&normsq[b * DIM + c], red[0]);
}

// ---------------- depthwise 3x3 for v, store transposed [bh][n][c_local] ----------------
// grid (HW/16, DIM/16, B2), block 256 = 16 channels x 16 pixels, LDS-staged transpose
__global__ void k_dwconv_v(const float* __restrict__ src, const float* __restrict__ w9,
                           _Float16* __restrict__ v_t) {
    int b = blockIdx.z, ct = blockIdx.y;          // 24 channel tiles of 16
    int n0 = blockIdx.x * 16;                     // 16 consecutive pixels (same row)
    int y = n0 >> 7, x0 = n0 & 127;
    int ci = threadIdx.x >> 4;                    // compute mapping: pixel fastest per channel
    int p  = threadIdx.x & 15;
    int c  = ct * 16 + ci;
    int x  = x0 + p;
    const float* sp = src + ((size_t)(b * DIM + c)) * HW;
    const float* wp = w9 + c * 9;
    float acc = 0.f;
    #pragma unroll
    for (int ky = 0; ky < 3; ++ky) {
        int yy = y + ky - 1;
        if (yy < 0 || yy >= HH) continue;
        #pragma unroll
        for (int kx = 0; kx < 3; ++kx) {
            int xx = x + kx - 1;
            if (xx < 0 || xx >= WW) continue;
            acc += sp[yy * WW + xx] * wp[ky * 3 + kx];
        }
    }
    __shared__ float lds[16][17];
    lds[ci][p] = acc;
    __syncthreads();
    // write mapping: channel fastest -> contiguous 32B stores into v_t
    int p2 = threadIdx.x >> 4;
    int c2 = threadIdx.x & 15;
    int head = ct / 3;
    int clb  = (ct % 3) * 16;                     // channel-in-head base
    int bh   = b * HEADS + head;
    v_t[(((size_t)bh * HW) + (n0 + p2)) * CP + clb + c2] = (_Float16)lds[c2][p2];
}

// ---------------- attn = q @ k^T  (raw dot products, split-K, f32 atomic reduce) ----------
// grid (32 splits, BH), block 32 (one wave); K chunk = 512 hw elements = 16 wmma steps
__global__ void k_attn_gemm(const _Float16* __restrict__ qh, const _Float16* __restrict__ kh,
                            float* __restrict__ attn) {
    int bh    = blockIdx.y;
    int split = blockIdx.x;
    int lane  = threadIdx.x & 31;
    const _Float16* qb = qh + (size_t)bh * CH * HW;
    const _Float16* kb = kh + (size_t)bh * CH * HW;
    int row   = lane & 15;
    int koffA = (lane < 16) ? 0 : 16;
    int koffB = (lane < 16) ? 0 : 16;
    v8f zero = {};
    v8f acc[3][3];
    #pragma unroll
    for (int m = 0; m < 3; ++m)
        #pragma unroll
        for (int t = 0; t < 3; ++t) acc[m][t] = zero;

    int nbase0 = split * 512;
    for (int ks = 0; ks < 16; ++ks) {
        int nb = nbase0 + ks * 32;
        F16x16 A[3], Bm[3];
        #pragma unroll
        for (int m = 0; m < 3; ++m) {
            const _Float16* pa = qb + (size_t)(m * 16 + row) * HW + nb + koffA;
            A[m].h[0] = *(const v8h*)pa;
            A[m].h[1] = *(const v8h*)(pa + 8);
        }
        #pragma unroll
        for (int t = 0; t < 3; ++t) {
            const _Float16* pb = kb + (size_t)(t * 16 + row) * HW + nb + koffB;
            Bm[t].v = *(const v16h*)pb;
        }
        #pragma unroll
        for (int m = 0; m < 3; ++m)
            #pragma unroll
            for (int t = 0; t < 3; ++t)
                acc[m][t] = __builtin_amdgcn_wmma_f32_16x16x32_f16(
                    false, A[m].v, false, Bm[t].v, (short)0, acc[m][t], false, false);
    }
    float* ab = attn + (size_t)bh * CH * CH;
    int rbase = (lane < 16) ? 0 : 8;
    #pragma unroll
    for (int m = 0; m < 3; ++m)
        #pragma unroll
        for (int t = 0; t < 3; ++t)
            #pragma unroll
            for (int r = 0; r < 8; ++r)
                atomicAdd(&ab[(m * 16 + rbase + r) * CH + t * 16 + row], acc[m][t][r]);
}

// ------- scale by inv norms * temperature, 4x top-k masked softmax, combine into M -------
// grid (BH), block 64 (48 active rows)
__global__ void k_topk_combine(const float* __restrict__ attn, const float* __restrict__ nq,
                               const float* __restrict__ nk, const float* __restrict__ temp,
                               const float* __restrict__ attnw, _Float16* __restrict__ Mm) {
    int bh = blockIdx.x;
    int c  = threadIdx.x;
    if (c >= CH) return;
    int head = bh & (HEADS - 1);
    float invq = 1.0f / fmaxf(sqrtf(nq[bh * CH + c]), 1e-12f);
    float t    = temp[head];
    const float* arow = attn + ((size_t)bh * CH + c) * CH;
    float a[CH], s[CH];
    for (int d = 0; d < CH; ++d) {
        float invk = 1.0f / fmaxf(sqrtf(nk[bh * CH + d]), 1e-12f);
        a[d] = arow[d] * invq * invk * t;
    }
    // insertion sort (descending) to get top-k thresholds
    for (int d = 0; d < CH; ++d) {
        float v = a[d];
        int j = d;
        while (j > 0 && s[j - 1] < v) { s[j] = s[j - 1]; --j; }
        s[j] = v;
    }
    const int ks[4] = { CH / 2, CH * 2 / 3, CH * 3 / 4, CH * 4 / 5 }; // 24,32,36,38
    float thr[4], den[4], w[4];
    float amax = s[0];
    #pragma unroll
    for (int i = 0; i < 4; ++i) { thr[i] = s[ks[i] - 1]; den[i] = 0.f; w[i] = attnw[i]; }
    for (int d = 0; d < CH; ++d) {
        float e = __expf(a[d] - amax);
        #pragma unroll
        for (int i = 0; i < 4; ++i) if (a[d] >= thr[i]) den[i] += e;
    }
    _Float16* mrow = Mm + ((size_t)bh * CH + c) * CP;
    for (int d = 0; d < CH; ++d) {
        float e = __expf(a[d] - amax);
        float m = 0.f;
        #pragma unroll
        for (int i = 0; i < 4; ++i) if (a[d] >= thr[i]) m += w[i] * e / den[i];
        mrow[d] = (_Float16)m;
    }
    for (int d = CH; d < CP; ++d) mrow[d] = (_Float16)0.f;   // K padding
}

// ---------------- out = M(48x64pad) @ v_t(64pad x HW), write out_t[b][n][c] f16 -----------
// grid (HW/16/4, BH), block 128 = 4 waves, one n-tile each
__global__ void k_out_gemm(const _Float16* __restrict__ Mm, const _Float16* __restrict__ v_t,
                           _Float16* __restrict__ out_t) {
    int bh    = blockIdx.y;
    int wave  = threadIdx.x >> 5;
    int lane  = threadIdx.x & 31;
    int ntile = blockIdx.x * 4 + wave;
    int row   = lane & 15;
    int koffA = (lane < 16) ? 0 : 16;
    int koffB = (lane < 16) ? 0 : 16;
    const _Float16* mb = Mm + (size_t)bh * CH * CP;
    const _Float16* vb = v_t + (size_t)bh * HW * CP;
    F16x16 A[3][2];
    #pragma unroll
    for (int m = 0; m < 3; ++m)
        #pragma unroll
        for (int k = 0; k < 2; ++k) {
            const _Float16* pa = mb + (m * 16 + row) * CP + k * 32 + koffA;
            A[m][k].h[0] = *(const v8h*)pa;
            A[m][k].h[1] = *(const v8h*)(pa + 8);
        }
    v8f zero = {};
    v8f acc[3] = { zero, zero, zero };
    int n = ntile * 16 + row;
    #pragma unroll
    for (int k = 0; k < 2; ++k) {
        F16x16 Bf;
        Bf.v = *(const v16h*)(vb + (size_t)n * CP + k * 32 + koffB);
        #pragma unroll
        for (int m = 0; m < 3; ++m)
            acc[m] = __builtin_amdgcn_wmma_f32_16x16x32_f16(
                false, A[m][k].v, false, Bf.v, (short)0, acc[m], false, false);
    }
    int b = bh >> 3, head = bh & 7;
    int rbase = (lane < 16) ? 0 : 8;
    #pragma unroll
    for (int m = 0; m < 3; ++m) {
        v8h o;
        #pragma unroll
        for (int r = 0; r < 8; ++r) o[r] = (_Float16)acc[m][r];
        *(v8h*)(out_t + ((size_t)b * HW + n) * DIM + head * CH + m * 16 + rbase) = o;
    }
}

// ---------------- final = Wproj(384x384) @ out_t, f32 NCHW output ----------------
// grid (HW/16/4, DIM/64, B2), block 128 = 4 waves; each wave: 4 m-tiles x 1 n-tile
// (4 m-tiles per wave -> B panel (out_t) read only 6x instead of 24x)
__global__ void k_proj_gemm(const _Float16* __restrict__ wp, const _Float16* __restrict__ out_t,
                            float* __restrict__ out) {
    int b    = blockIdx.z, mg = blockIdx.y;       // m-group of 4 tiles
    int wave = threadIdx.x >> 5;
    int lane = threadIdx.x & 31;
    int ntile = blockIdx.x * 4 + wave;
    int row   = lane & 15;
    int koffA = (lane < 16) ? 0 : 16;
    int koffB = (lane < 16) ? 0 : 16;
    int n = ntile * 16 + row;
    const _Float16* bb = out_t + ((size_t)b * HW + n) * DIM;
    v8f zero = {};
    v8f acc[4] = { zero, zero, zero, zero };
    for (int k = 0; k < DIM / 32; ++k) {
        F16x16 Bf;
        Bf.v = *(const v16h*)(bb + k * 32 + koffB);
        F16x16 A[4];
        #pragma unroll
        for (int mi = 0; mi < 4; ++mi) {
            const _Float16* pa = wp + ((mg * 4 + mi) * 16 + row) * DIM + k * 32 + koffA;
            A[mi].h[0] = *(const v8h*)pa;
            A[mi].h[1] = *(const v8h*)(pa + 8);
        }
        #pragma unroll
        for (int mi = 0; mi < 4; ++mi)
            acc[mi] = __builtin_amdgcn_wmma_f32_16x16x32_f16(
                false, A[mi].v, false, Bf.v, (short)0, acc[mi], false, false);
    }
    int rbase = (lane < 16) ? 0 : 8;
    #pragma unroll
    for (int mi = 0; mi < 4; ++mi)
        #pragma unroll
        for (int r = 0; r < 8; ++r)
            out[((size_t)(b * DIM) + (mg * 4 + mi) * 16 + rbase + r) * HW + n] = acc[mi][r];
}

extern "C" void kernel_launch(void* const* d_in, const int* in_sizes, int n_in,
                              void* d_out, int out_size, void* d_ws, size_t ws_size,
                              hipStream_t stream) {
    const float* k_fea = (const float*)d_in[0];
    const float* v_fea = (const float*)d_in[1];
    const float* q_fea = (const float*)d_in[2];
    const float* wq    = (const float*)d_in[3];
    const float* wk    = (const float*)d_in[4];
    const float* wv    = (const float*)d_in[5];
    const float* wproj = (const float*)d_in[6];
    const float* temp  = (const float*)d_in[7];
    const float* attnw = (const float*)d_in[8];
    float* out = (float*)d_out;
    char* ws = (char*)d_ws;
    if (ws_size < WS_NEED) return;

    _Float16* qh    = (_Float16*)(ws + OFF_QH);
    _Float16* kh    = (_Float16*)(ws + OFF_KH);
    _Float16* v_t   = (_Float16*)(ws + OFF_VT);
    _Float16* out_t = (_Float16*)(ws + OFF_OUTT);
    float*    attn  = (float*)(ws + OFF_ATTN);
    _Float16* Mm    = (_Float16*)(ws + OFF_M);
    float*    nq    = (float*)(ws + OFF_NQ);
    float*    nk    = (float*)(ws + OFF_NK);
    _Float16* wp16  = (_Float16*)(ws + OFF_WP);

    // zero accumulators & padded v
    hipMemsetAsync(attn, 0, (size_t)BH * CH * CH * 4, stream);
    hipMemsetAsync(nq, 0, 768 * 4, stream);
    hipMemsetAsync(nk, 0, 768 * 4, stream);
    hipMemsetAsync(v_t, 0, (size_t)BH * HW * CP * 2, stream);

    k_cvt_wproj<<<dim3((DIM * DIM) / 256), 256, 0, stream>>>(wproj, wp16);

    dim3 gqk(HW / 256, DIM, B2);
    k_dwconv_qk<<<gqk, 256, 0, stream>>>(q_fea, wq, qh, nq);
    k_dwconv_qk<<<gqk, 256, 0, stream>>>(k_fea, wk, kh, nk);

    k_dwconv_v<<<dim3(HW / 16, DIM / 16, B2), 256, 0, stream>>>(v_fea, wv, v_t);

    k_attn_gemm<<<dim3(32, BH), 32, 0, stream>>>(qh, kh, attn);

    k_topk_combine<<<dim3(BH), 64, 0, stream>>>(attn, nq, nk, temp, attnw, Mm);

    k_out_gemm<<<dim3(HW / 64, BH), 128, 0, stream>>>(Mm, v_t, out_t);

    k_proj_gemm<<<dim3(HW / 64, DIM / 64, B2), 128, 0, stream>>>(wp16, out_t, out);
}
